// AttentionOnDetail_3968549781612
// MI455X (gfx1250) — matrix operand order repaired and checked by reading, not compiled, not run on hardware
//
#include <hip/hip_runtime.h>
#include <hip/hip_bf16.h>

// ---------------- problem constants ----------------
constexpr int   kB  = 2;
constexpr int   kT  = 2048;
constexpr int   kC  = 1024;
constexpr int   kH  = 16;
constexpr int   kHD = 64;
constexpr int   kM  = kB * kT;          // 4096 rows for the projections
constexpr float kEPS   = 1.1920929e-07f;
constexpr float kSCALE = 0.12f;
constexpr float kPI    = 3.14159265358979323846f;

typedef __attribute__((ext_vector_type(16))) __bf16        v16bf;
typedef __attribute__((ext_vector_type(8)))  float         v8f;
typedef __attribute__((ext_vector_type(4)))  unsigned int  v4u;
typedef __attribute__((ext_vector_type(4)))  int           v4i;
typedef __attribute__((ext_vector_type(8)))  int           v8i;

union FragU { uint4 u[2]; v16bf b; };

#ifndef __has_builtin
#define __has_builtin(x) 0
#endif
#if defined(__AMDGCN__) && __has_builtin(__builtin_amdgcn_tensor_load_to_lds)
#define ATH_TDM 1
#else
#define ATH_TDM 0
#endif

// ---------------- small helpers ----------------
__device__ __forceinline__ unsigned short f2bf(float x) {
    unsigned int u = __float_as_uint(x);
    u += 0x7fffu + ((u >> 16) & 1u);          // round-to-nearest-even
    return (unsigned short)(u >> 16);
}

__device__ __forceinline__ float redsum16(float v) {
    v += __shfl_xor(v, 1);  v += __shfl_xor(v, 2);
    v += __shfl_xor(v, 4);  v += __shfl_xor(v, 8);
    return v;               // reduction stays inside each 16-lane half of the wave32
}
__device__ __forceinline__ float redmax16(float v) {
    v = fmaxf(v, __shfl_xor(v, 1));  v = fmaxf(v, __shfl_xor(v, 2));
    v = fmaxf(v, __shfl_xor(v, 4));  v = fmaxf(v, __shfl_xor(v, 8));
    return v;
}

// RMS norm over one head (64 channels = 16 threads x 4 channels).
__device__ __forceinline__ void rms4(float v[4]) {
    float ss = v[0]*v[0] + v[1]*v[1] + v[2]*v[2] + v[3]*v[3];
    ss = redsum16(ss);
    float r = rsqrtf(ss * (1.0f / 64.0f) + kEPS);
    v[0] *= r; v[1] *= r; v[2] *= r; v[3] *= r;
}

// Rotary for 4 consecutive channels. ht = thread index within the head (0..15).
// d0 = 4*ht; pair (d, d+32) lives in threads (ht, ht+8) -> lane^8 shuffle.
__device__ __forceinline__ void rot4(float v[4], int tpos, int ht) {
    bool first = (ht & 8) == 0;
    #pragma unroll
    for (int j = 0; j < 4; ++j) {
        float other = __shfl_xor(v[j], 8);
        int idx = ((ht & 7) << 2) + j;        // freq index 0..31
        float cs = 1.0f, sn = 0.0f;
        if (idx < 16) {                        // idx>=16 -> freq 0 -> identity
            float fr = exp2f((float)idx * (-10.0f / 15.0f));  // (1/1024)^(idx/15)
            __sincosf((float)tpos * fr, &sn, &cs);
        }
        v[j] = first ? (v[j] * cs + other * sn) : (v[j] * cs - other * sn);
    }
}

#if ATH_TDM
// Issue a TDM 2D tile load: 128 rows x 32 bf16 from a row-major matrix with
// row stride `stride0` elements, into LDS with 16B padding after each 64B row
// (-> 80B row pitch, matching the stride-40-ushort fragment layout).
// D# bitfields per CDNA5 ISA ch.8 (group0: count/lds/global/type,
// group1: data_size, pad, dims, strides).
__device__ __forceinline__ void tdm_load_tile(unsigned lds_addr,
                                              const unsigned short* gptr,
                                              int tensor_d0, int tensor_d1,
                                              int stride0) {
    unsigned long long ga = (unsigned long long)(uintptr_t)gptr;
    v4u g0 = { 1u,                                    // count=1, user descriptor
               lds_addr,                              // bits 63:32 lds_addr
               (unsigned)(ga & 0xffffffffu),          // global_addr lo
               (unsigned)((ga >> 32) & 0x01ffffffu) | (2u << 30) }; // addr hi | type=2
    v8i g1;
    g1[0] = (1 << 16)       // data_size = 2 bytes
          | (1 << 20)       // pad_enable
          | (3 << 22)       // pad_interval: 16 DWORDs (=64B) between pads
          | (3 << 25);      // pad_amount: 4 DWORDs (=16B)
    g1[1] = (int)(((unsigned)tensor_d0 & 0xffffu) << 16);              // dim0[15:0]
    g1[2] = (int)((((unsigned)tensor_d0 >> 16) & 0xffffu)
          |      (((unsigned)tensor_d1 & 0xffffu) << 16));             // dim0 hi | dim1 lo
    g1[3] = (int)((((unsigned)tensor_d1 >> 16) & 0xffffu)
          |      (32u << 16));                                         // dim1 hi | tile_dim0=32
    g1[4] = 128;            // tile_dim1 = 128 rows (tile_dim2 = 0)
    g1[5] = stride0;        // tensor_dim0_stride lo (elements)
    g1[6] = 0;
    g1[7] = 0;
    v4i z4 = {0, 0, 0, 0};
#if __clang_major__ >= 23
    v8i z8 = {0, 0, 0, 0, 0, 0, 0, 0};
    __builtin_amdgcn_tensor_load_to_lds(g0, g1, z4, z4, z8, 0);
#else
    __builtin_amdgcn_tensor_load_to_lds(g0, g1, z4, z4, 0);
#endif
}
__device__ __forceinline__ unsigned lds_off(const void* p) {
    return (unsigned)(uintptr_t)p;    // generic addr low 32 bits = LDS byte offset
}
#endif

// ---------------- kernel 1: split c_proj into bf16 weight halves ----------------
__global__ __launch_bounds__(256) void conv_w(const float* __restrict__ cp,
                                              unsigned short* __restrict__ aw,
                                              unsigned short* __restrict__ mw) {
    int idx = blockIdx.x * 256 + threadIdx.x;         // kC*kC/4 threads
    int e  = idx * 4;
    int n  = e / kC, k = e % kC;
    const float4 a = *(const float4*)(cp + (size_t)n * 2 * kC + k);
    const float4 m = *(const float4*)(cp + (size_t)n * 2 * kC + kC + k);
    uint2 pa, pm;
    pa.x = (unsigned)f2bf(a.x) | ((unsigned)f2bf(a.y) << 16);
    pa.y = (unsigned)f2bf(a.z) | ((unsigned)f2bf(a.w) << 16);
    pm.x = (unsigned)f2bf(m.x) | ((unsigned)f2bf(m.y) << 16);
    pm.y = (unsigned)f2bf(m.z) | ((unsigned)f2bf(m.w) << 16);
    *(uint2*)(aw + (size_t)n * kC + k) = pa;
    *(uint2*)(mw + (size_t)n * kC + k) = pm;
}

// ---------------- kernel 2: AFT front (harmonics + RMS + rotary + exp) ----------------
__global__ __launch_bounds__(256) void proc_aft(const float* __restrict__ x,
                                                const float* __restrict__ kqv,
                                                float* __restrict__ sigq,
                                                float* __restrict__ ekb,
                                                float* __restrict__ vnb) {
    __shared__ float kq[30];
    int tid = threadIdx.x;
    if (tid < 30) kq[tid] = kqv[tid];
    __syncthreads();

    int bid = blockIdx.x;
    int b = bid / kT, t = bid % kT;
    int c0 = tid * 4;
    int h = c0 >> 6, ht = tid & 15;

    const float4 xv = *(const float4*)(x + ((size_t)b * kT + t) * kC + c0);
    float xi[4] = {xv.x, xv.y, xv.z, xv.w};
    float q[4], k[4], v[4];
    #pragma unroll
    for (int j = 0; j < 4; ++j) {
        float sg = 1.0f / (1.0f + __expf(-xi[j]));
        float xs = 2.0f * kPI * sg - kPI;
        float s1, c1; __sincosf(xs, &s1, &c1);
        float s2 = 2.0f * s1 * c1,  c2 = c1 * c1 - s1 * s1;
        float s3 = s1 * c2 + c1 * s2, c3 = c1 * c2 - s1 * s2;
        float s4 = 2.0f * s2 * c2,  c4 = c2 * c2 - s2 * s2;
        float r[3];
        #pragma unroll
        for (int i = 0; i < 3; ++i) {
            const float* kr = kq + 10 * i;
            r[i] = kr[0] + kr[5]
                 + kr[1] * s1 + kr[2] * s2 + kr[3] * s3 + kr[4] * s4
                 + kr[6] * c1 + kr[7] * c2 + kr[8] * c3 + kr[9] * c4;
        }
        q[j] = r[0]; k[j] = r[1]; v[j] = r[2];
    }
    rms4(q); rms4(k); rms4(v);
    rot4(q, t, ht); rot4(k, t, ht);

    size_t ob = (((size_t)b * kH + h) * kT + t) * kHD + (c0 & 63);
    float4 oq = {1.0f/(1.0f+__expf(-q[0])), 1.0f/(1.0f+__expf(-q[1])),
                 1.0f/(1.0f+__expf(-q[2])), 1.0f/(1.0f+__expf(-q[3]))};
    float4 oe = {__expf(k[0]), __expf(k[1]), __expf(k[2]), __expf(k[3])};
    float4 ov = {v[0], v[1], v[2], v[3]};
    *(float4*)(sigq + ob) = oq;
    *(float4*)(ekb  + ob) = oe;
    *(float4*)(vnb  + ob) = ov;
}

// ---------------- kernel 3: AFT combine (batch-sum) -> bf16 (B*T, C) ----------------
__global__ __launch_bounds__(256) void aft_combine(const float* __restrict__ sigq,
                                                   const float* __restrict__ ekb,
                                                   const float* __restrict__ vnb,
                                                   unsigned short* __restrict__ ybf) {
    int idx = blockIdx.x * 256 + threadIdx.x;   // B*T*C/4 threads
    int e  = idx * 4;
    int c0 = e % kC;
    int bt = e / kC;
    int t  = bt % kT, b = bt / kT;
    int h  = c0 >> 6, d0 = c0 & 63;
    size_t b0 = (((size_t)0 * kH + h) * kT + t) * kHD + d0;
    size_t b1 = (((size_t)1 * kH + h) * kT + t) * kHD + d0;
    size_t bq = (((size_t)b * kH + h) * kT + t) * kHD + d0;
    float4 sq = *(const float4*)(sigq + bq);
    float4 e0 = *(const float4*)(ekb + b0), e1 = *(const float4*)(ekb + b1);
    float4 v0 = *(const float4*)(vnb + b0), v1 = *(const float4*)(vnb + b1);
    float y0 = sq.x * (e0.x * v0.x + e1.x * v1.x) / (e0.x + e1.x);
    float y1 = sq.y * (e0.y * v0.y + e1.y * v1.y) / (e0.y + e1.y);
    float y2 = sq.z * (e0.z * v0.z + e1.z * v1.z) / (e0.z + e1.z);
    float y3 = sq.w * (e0.w * v0.w + e1.w * v1.w) / (e0.w + e1.w);
    uint2 o;
    o.x = (unsigned)f2bf(y0) | ((unsigned)f2bf(y1) << 16);
    o.y = (unsigned)f2bf(y2) | ((unsigned)f2bf(y3) << 16);
    *(uint2*)(ybf + ((size_t)b * kT + t) * kC + c0) = o;
}

// ---------------- kernel 4: bf16 GEMM  D[m,n] = sum_k A[m,k]*W[n,k]  (f32 out) -------
// 128x128 tile / block, BK=32, 8 waves (2x4), each wave 64x32 = 4x2 WMMA tiles.
// Staging: TDM tensor_load_to_lds, double-buffered (fallback: vector copy loop).
__global__ __launch_bounds__(256) void gemm_bf16(const unsigned short* __restrict__ A,
                                                 const unsigned short* __restrict__ W,
                                                 float* __restrict__ D,
                                                 int Mdim, int Ndim, int Kdim) {
    __shared__ __attribute__((aligned(16))) unsigned short As[2][128 * 40];
    __shared__ __attribute__((aligned(16))) unsigned short Ws[2][128 * 40];

    int tid  = threadIdx.x;
    int lane = tid & 31, wid = tid >> 5;
    int lh = lane & 15, hig = lane >> 4;              // half-wave select
    int row0 = blockIdx.y * 128, col0 = blockIdx.x * 128;
    int wm = (wid & 1) * 64, wn = (wid >> 1) * 32;

    const v8f vz = {0.f, 0.f, 0.f, 0.f, 0.f, 0.f, 0.f, 0.f};
    v8f acc[4][2];
    #pragma unroll
    for (int i = 0; i < 4; ++i)
        #pragma unroll
        for (int j = 0; j < 2; ++j) acc[i][j] = vz;

    const int nIter = Kdim / 32;

#if ATH_TDM
    if (wid == 0) {   // one wave drives the DMA; TENSORcnt is per-wave & in-order
        tdm_load_tile(lds_off(&As[0][0]), A + (size_t)row0 * Kdim, Kdim, Mdim - row0, Kdim);
        tdm_load_tile(lds_off(&Ws[0][0]), W + (size_t)col0 * Kdim, Kdim, Ndim - col0, Kdim);
    }
#endif

    for (int it = 0; it < nIter; ++it) {
        int buf = it & 1;
#if ATH_TDM
        if (wid == 0) {
            if (it + 1 < nIter) {   // prefetch next tile pair into the other buffer
                tdm_load_tile(lds_off(&As[buf ^ 1][0]),
                              A + (size_t)row0 * Kdim + (it + 1) * 32,
                              Kdim, Mdim - row0, Kdim);
                tdm_load_tile(lds_off(&Ws[buf ^ 1][0]),
                              W + (size_t)col0 * Kdim + (it + 1) * 32,
                              Kdim, Ndim - col0, Kdim);
                __builtin_amdgcn_s_wait_tensorcnt(2);  // current pair has landed
            } else {
                __builtin_amdgcn_s_wait_tensorcnt(0);
            }
        }
        __syncthreads();                                // publish LDS tile to all waves
#else
        __syncthreads();
        #pragma unroll
        for (int i = 0; i < 2; ++i) {                   // 512 16B chunks, 2/thread
            int idx = tid + i * 256;
            int r = idx >> 2, cc = (idx & 3) * 8;
            *(uint4*)(&As[buf][0] + r * 40 + cc) =
                *(const uint4*)(A + (size_t)(row0 + r) * Kdim + it * 32 + cc);
            *(uint4*)(&Ws[buf][0] + r * 40 + cc) =
                *(const uint4*)(W + (size_t)(col0 + r) * Kdim + it * 32 + cc);
        }
        __syncthreads();
#endif

        FragU af[4], bfr[2];
        #pragma unroll
        for (int i = 0; i < 4; ++i) {                 // A layout: K {0..7,16..23}/{8..15,24..31}
            const unsigned short* p = &As[buf][0] + (wm + 16 * i + lh) * 40;
            af[i].u[0] = *(const uint4*)(p + hig * 8);
            af[i].u[1] = *(const uint4*)(p + 16 + hig * 8);
        }
        #pragma unroll
        for (int j = 0; j < 2; ++j) {                 // B layout: K {0..7,16..23}/{8..15,24..31}
            const unsigned short* p = &Ws[buf][0] + (wn + 16 * j + lh) * 40;
            bfr[j].u[0] = *(const uint4*)(p + hig * 8);
            bfr[j].u[1] = *(const uint4*)(p + 16 + hig * 8);
        }
        #pragma unroll
        for (int i = 0; i < 4; ++i)
            #pragma unroll
            for (int j = 0; j < 2; ++j)
                acc[i][j] = __builtin_amdgcn_wmma_f32_16x16x32_bf16(
                    false, af[i].b, false, bfr[j].b, (short)0, acc[i][j], false, false);

#if ATH_TDM
        __syncthreads();   // all reads done before wave0 re-targets this buffer
#endif
    }

    #pragma unroll
    for (int i = 0; i < 4; ++i)
        #pragma unroll
        for (int j = 0; j < 2; ++j)
            #pragma unroll
            for (int r = 0; r < 8; ++r) {
                int m = row0 + wm + 16 * i + r + hig * 8;
                int n = col0 + wn + 16 * j + lh;
                D[(size_t)m * Ndim + n] = acc[i][j][r];
            }
}

// ---------------- kernel 5: MHA prep (RMS + rotary on a; q==k by construction) -------
__global__ __launch_bounds__(256) void proc_mha(const float* __restrict__ a,
                                                unsigned short* __restrict__ q2,
                                                unsigned short* __restrict__ v2) {
    int tid = threadIdx.x;
    int bid = blockIdx.x;
    int b = bid / kT, t = bid % kT;
    int c0 = tid * 4;
    int h = c0 >> 6, ht = tid & 15;
    float4 av = *(const float4*)(a + ((size_t)b * kT + t) * kC + c0);
    float nv[4] = {av.x, av.y, av.z, av.w};
    rms4(nv);
    float qv[4] = {nv[0], nv[1], nv[2], nv[3]};
    rot4(qv, t, ht);
    size_t ob = (((size_t)b * kH + h) * kT + t) * kHD + (c0 & 63);
    uint2 pq, pv;
    pq.x = (unsigned)f2bf(qv[0]) | ((unsigned)f2bf(qv[1]) << 16);
    pq.y = (unsigned)f2bf(qv[2]) | ((unsigned)f2bf(qv[3]) << 16);
    pv.x = (unsigned)f2bf(nv[0]) | ((unsigned)f2bf(nv[1]) << 16);
    pv.y = (unsigned)f2bf(nv[2]) | ((unsigned)f2bf(nv[3]) << 16);
    *(uint2*)(q2 + ob) = pq;
    *(uint2*)(v2 + ob) = pv;
}

// ---------------- kernel 6: causal flash attention with WMMA -------------------------
// Block: 128 threads (4 waves) = 64 query rows for one (b,h). Key blocks of 32.
__global__ __launch_bounds__(128) void attn(const unsigned short* __restrict__ q2,
                                            const unsigned short* __restrict__ v2,
                                            unsigned short* __restrict__ av) {
    __shared__ __attribute__((aligned(16))) unsigned short Ks[32 * 80];     // [key][hd]
    __shared__ __attribute__((aligned(16))) unsigned short Vt[64 * 40];     // [hd][key]
    __shared__ __attribute__((aligned(16))) unsigned short Ps[4 * 16 * 40]; // per-wave P

    int tid  = threadIdx.x;
    int lane = tid & 31, w = tid >> 5;
    int lh = lane & 15, hig = lane >> 4;
    int nqb  = kT / 64;
    int qblk = blockIdx.x % nqb;
    int bh   = blockIdx.x / nqb;
    int b = bh / kH, h = bh % kH;
    int qb = qblk * 64;
    int qrow0 = qb + 16 * w;

    const unsigned short* qh = q2 + (size_t)bh * kT * kHD;   // K == Q here
    const unsigned short* vh = v2 + (size_t)bh * kT * kHD;

    // preload this wave's Q tile (16 x 64) as two A-fragments
    FragU qf[2];
    {
        const unsigned short* p = qh + (size_t)(qrow0 + lh) * kHD;
        qf[0].u[0] = *(const uint4*)(p + hig * 8);
        qf[0].u[1] = *(const uint4*)(p + 16 + hig * 8);
        qf[1].u[0] = *(const uint4*)(p + 32 + hig * 8);
        qf[1].u[1] = *(const uint4*)(p + 48 + hig * 8);
    }

    const v8f vz = {0.f, 0.f, 0.f, 0.f, 0.f, 0.f, 0.f, 0.f};
    v8f oac[4];
    #pragma unroll
    for (int j = 0; j < 4; ++j) oac[j] = vz;
    float mrow[8], lrow[8];
    #pragma unroll
    for (int r = 0; r < 8; ++r) { mrow[r] = -3.0e38f; lrow[r] = 0.0f; }

    unsigned short* Pw = Ps + w * 16 * 40;
    int nkb = qb / 32 + 2;                       // causal: keys up to qb+63

    for (int kb = 0; kb < nkb; ++kb) {
        int kb0 = kb * 32;
        // hint the next key block toward the caches (global_prefetch_b8)
        __builtin_prefetch(qh + (size_t)(kb0 + 32) * kHD + lane * 16, 0, 1);
        __builtin_prefetch(vh + (size_t)(kb0 + 32) * kHD + lane * 16, 0, 1);
        __syncthreads();
        #pragma unroll
        for (int i = 0; i < 2; ++i) {            // stage K (row-major) and V (transposed)
            int idx = tid + i * 128;
            int r = idx >> 3, cc = (idx & 7) * 8;
            *(uint4*)(Ks + r * 80 + cc) =
                *(const uint4*)(qh + (size_t)(kb0 + r) * kHD + cc);
            uint4 gv = *(const uint4*)(vh + (size_t)(kb0 + r) * kHD + cc);
            const unsigned short* pv = (const unsigned short*)&gv;
            #pragma unroll
            for (int e = 0; e < 8; ++e) Vt[(cc + e) * 40 + r] = pv[e];
        }
        __syncthreads();

        // S = Q Kt : two 16x16 tiles, each 2 WMMAs over hd
        v8f s[2];
        #pragma unroll
        for (int nt = 0; nt < 2; ++nt) {
            v8f sv = vz;
            #pragma unroll
            for (int kh = 0; kh < 2; ++kh) {
                FragU kf;
                const unsigned short* p = Ks + (nt * 16 + lh) * 80 + kh * 32;
                kf.u[0] = *(const uint4*)(p + hig * 8);
                kf.u[1] = *(const uint4*)(p + 16 + hig * 8);
                sv = __builtin_amdgcn_wmma_f32_16x16x32_bf16(
                    false, qf[kh].b, false, kf.b, (short)0, sv, false, false);
            }
            s[nt] = sv;
        }

        // online softmax per row (rows live on the 16-lane half, slot r)
        int key0 = kb0 + lh, key1 = kb0 + 16 + lh;
        #pragma unroll
        for (int r = 0; r < 8; ++r) {
            int trow = qrow0 + r + hig * 8;
            float s0 = s[0][r] * kSCALE; if (key0 > trow) s0 = -3.0e38f;
            float s1 = s[1][r] * kSCALE; if (key1 > trow) s1 = -3.0e38f;
            float mx   = redmax16(fmaxf(s0, s1));
            float mnew = fmaxf(mrow[r], mx);
            float corr = __expf(mrow[r] - mnew);
            float p0 = (s0 < -1.0e30f) ? 0.0f : __expf(s0 - mnew);
            float p1 = (s1 < -1.0e30f) ? 0.0f : __expf(s1 - mnew);
            float rs = redsum16(p0 + p1);
            lrow[r] = lrow[r] * corr + rs;
            mrow[r] = mnew;
            #pragma unroll
            for (int j = 0; j < 4; ++j) oac[j][r] *= corr;
            Pw[(r + hig * 8) * 40 + lh]      = f2bf(p0);
            Pw[(r + hig * 8) * 40 + 16 + lh] = f2bf(p1);
        }
        asm volatile("s_wait_dscnt 0" ::: "memory");   // wave-local P turnaround

        // O += P V : reload P as A-fragment, V columns contiguous in Vt
        FragU pf;
        {
            const unsigned short* p = Pw + lh * 40;
            pf.u[0] = *(const uint4*)(p + hig * 8);
            pf.u[1] = *(const uint4*)(p + 16 + hig * 8);
        }
        #pragma unroll
        for (int j = 0; j < 4; ++j) {
            FragU vf;
            const unsigned short* p = Vt + (16 * j + lh) * 40;
            vf.u[0] = *(const uint4*)(p + hig * 8);
            vf.u[1] = *(const uint4*)(p + 16 + hig * 8);
            oac[j] = __builtin_amdgcn_wmma_f32_16x16x32_bf16(
                false, pf.b, false, vf.b, (short)0, oac[j], false, false);
        }
    }

    // write O/l as bf16 directly into (B,T,C) layout for the final projection
    #pragma unroll
    for (int j = 0; j < 4; ++j)
        #pragma unroll
        for (int r = 0; r < 8; ++r) {
            int trow = qrow0 + r + hig * 8;
            int cc = h * 64 + 16 * j + lh;
            float val = oac[j][r] / lrow[r];
            av[((size_t)b * kT + trow) * kC + cc] = f2bf(val);
        }
}

// ---------------- host-side orchestration ----------------
extern "C" void kernel_launch(void* const* d_in, const int* in_sizes, int n_in,
                              void* d_out, int out_size, void* d_ws, size_t ws_size,
                              hipStream_t stream) {
    (void)in_sizes; (void)n_in; (void)out_size; (void)ws_size;
    const float* x      = (const float*)d_in[0];
    const float* kqv    = (const float*)d_in[1];
    const float* c_proj = (const float*)d_in[2];
    float* out = (float*)d_out;

    char* ws = (char*)d_ws;
    const size_t SZ_BHTD_F  = (size_t)kB * kH * kT * kHD * 4;  // 16 MB
    const size_t SZ_BTC_F   = (size_t)kB * kT * kC * 4;        // 16 MB
    const size_t SZ_BTC_BF  = (size_t)kB * kT * kC * 2;        //  8 MB
    const size_t SZ_W_BF    = (size_t)kC * kC * 2;             //  2 MB

    size_t off = 0;
    float*          sigq = (float*)(ws + off);          off += SZ_BHTD_F;
    float*          ekb  = (float*)(ws + off);          off += SZ_BHTD_F;
    float*          vnb  = (float*)(ws + off);          off += SZ_BHTD_F;
    float*          abuf = (float*)(ws + off);          off += SZ_BTC_F;
    unsigned short* ybf  = (unsigned short*)(ws + off); off += SZ_BTC_BF;
    unsigned short* q2   = (unsigned short*)(ws + off); off += SZ_BTC_BF;
    unsigned short* v2   = (unsigned short*)(ws + off); off += SZ_BTC_BF;
    unsigned short* avb  = (unsigned short*)(ws + off); off += SZ_BTC_BF;
    unsigned short* aw   = (unsigned short*)(ws + off); off += SZ_W_BF;
    unsigned short* mw   = (unsigned short*)(ws + off); off += SZ_W_BF;

    conv_w     <<<dim3(kC * kC / 4 / 256), dim3(256), 0, stream>>>(c_proj, aw, mw);
    proc_aft   <<<dim3(kB * kT),           dim3(256), 0, stream>>>(x, kqv, sigq, ekb, vnb);
    aft_combine<<<dim3(kB * kT * kC / 4 / 256), dim3(256), 0, stream>>>(sigq, ekb, vnb, ybf);
    gemm_bf16  <<<dim3(kC / 128, kM / 128), dim3(256), 0, stream>>>(ybf, aw, abuf, kM, kC, kC);
    proc_mha   <<<dim3(kB * kT),           dim3(256), 0, stream>>>(abuf, q2, v2);
    attn       <<<dim3(kB * kH * (kT / 64)), dim3(128), 0, stream>>>(q2, v2, avb);
    gemm_bf16  <<<dim3(kC / 128, kM / 128), dim3(256), 0, stream>>>(avb, mw, out, kM, kC, kC);
}
